// WordRNN_87514253623880
// MI455X (gfx1250) — matrix operand order repaired and checked by reading, not compiled, not run on hardware
//
#include <hip/hip_runtime.h>

// ---------------------------------------------------------------------------
// MI455X (gfx1250) fused 2-layer LSTM + dense softmax, wave32 WMMA f16.
//
// Design: 32 workgroups (one per 16-row batch tile) x 256 threads (8 waves).
// Wave w owns hidden slice [16w,16w+16): its 4 z-column tiles are exactly the
// i/f/g/o gates for that slice, so gate math + cell state stay in registers.
// h state is shared via LDS in f16; weights are pre-packed into WMMA
// B-fragment layout in d_ws (compiler hoists the loop-invariant fragments
// into registers across the T loop, per round-1 asm).
// Dense head: [16,64]x[64,32000] per workgroup, 250 N-tiles per wave, with
// ONLINE softmax (running max + rescaled sum) -> only 2 passes over the 65MB
// output: 195MB HBM traffic total (~8us at 23.3 TB/s).
// ---------------------------------------------------------------------------

typedef __attribute__((ext_vector_type(16))) _Float16 v16h;
typedef __attribute__((ext_vector_type(8)))  _Float16 v8h;
typedef __attribute__((ext_vector_type(8)))  float    v8f;

#define TLEN  512
#define EDIM  32
#define H1D   128
#define H2D   64
#define VDICT 32000
#define EPSB  1e-3f

__device__ __forceinline__ v8f wmma_f16(v16h a, v16h b, v8f c) {
  // D = A(16x32 f16) * B(32x16 f16) + C(16x16 f32)
  return __builtin_amdgcn_wmma_f32_16x16x32_f16(false, a, false, b,
                                                (short)0, c, false, false);
}

__device__ __forceinline__ v16h cat8(v8h lo, v8h hi) {
  return __builtin_shufflevector(lo, hi, 0,1,2,3,4,5,6,7,8,9,10,11,12,13,14,15);
}

__device__ __forceinline__ v8f splat8(float s) {
  v8f r;
#pragma unroll
  for (int i = 0; i < 8; ++i) r[i] = s;
  return r;
}

__device__ __forceinline__ float sigmoidf(float x) {
  return 1.0f / (1.0f + __expf(-x));
}

// A-fragment (16x32 f16) from row-major [16 x rowStride] f16 array.
// Lane layout: lanes 0-15 hold row M = lane, K = {0..7,16..23};
//              lanes 16-31 hold same rows, K = {8..15,24..31}.
__device__ __forceinline__ v16h load_a(const _Float16* base, int rowStride,
                                       int row, int kbase, int lane) {
  const _Float16* p = base + row * rowStride + kbase + ((lane < 16) ? 0 : 8);
  v8h lo = *(const v8h*)p;
  v8h hi = *(const v8h*)(p + 16);
  return cat8(lo, hi);
}

// B-fragment (32x16 f16): lane holds column N=lane&15, 16 contiguous halves
// covering K = (lane<16 ? 0..7,16..23 : 8..15,24..31) of the chunk. One 32B record/lane.
__device__ __forceinline__ v16h load_b(const _Float16* frag, int tile,
                                       int chunk, int nCh, int lane) {
  return *(const v16h*)(frag + (((tile * nCh + chunk) * 32) + lane) * 16);
}

// ---------------------------------------------------------------------------
// Prep kernels: f32 -> f16 conversion + WMMA B-fragment packing.
// ---------------------------------------------------------------------------
__global__ void cvt_half_kernel(const float* __restrict__ src,
                                _Float16* __restrict__ dst, int n) {
  int i = blockIdx.x * blockDim.x + threadIdx.x;
  if (i < n) dst[i] = (_Float16)src[i];
}

// Pack W [Ktot x Ntot] (row-major f32) into B-fragments.
__global__ void pack_b_kernel(const float* __restrict__ W,
                              _Float16* __restrict__ dst,
                              int nTiles, int nCh, int Ntot) {
  int i = blockIdx.x * blockDim.x + threadIdx.x;
  int total = nTiles * nCh * 32;
  if (i >= total) return;
  int lane  = i & 31;
  int tc    = i >> 5;
  int chunk = tc % nCh;
  int tile  = tc / nCh;
  int kb    = chunk * 32 + ((lane < 16) ? 0 : 8);
  int colg  = tile * 16 + (lane & 15);
  _Float16* o = dst + (size_t)i * 16;
#pragma unroll
  for (int h = 0; h < 16; ++h)
    o[h] = (_Float16)W[(size_t)(kb + ((h < 8) ? h : h + 8)) * Ntot + colg];
}

// ---------------------------------------------------------------------------
// Fused LSTM1 -> BN1 -> LSTM2 -> BN2 -> Dense -> online Softmax
// ---------------------------------------------------------------------------
__global__ __launch_bounds__(256)
void lstm_fused_kernel(const int* __restrict__ ids,
                       const _Float16* __restrict__ emb16,
                       const _Float16* __restrict__ w1f,
                       const _Float16* __restrict__ u1f,
                       const _Float16* __restrict__ w2f,
                       const _Float16* __restrict__ u2f,
                       const _Float16* __restrict__ wdf,
                       const float* __restrict__ b1,
                       const float* __restrict__ g1, const float* __restrict__ beta1,
                       const float* __restrict__ m1, const float* __restrict__ v1,
                       const float* __restrict__ b2,
                       const float* __restrict__ g2, const float* __restrict__ beta2,
                       const float* __restrict__ m2, const float* __restrict__ v2,
                       const float* __restrict__ bd,
                       float* __restrict__ out) {
  __shared__ _Float16 sh1raw[16 * H1D];   // h1_t (raw), f16
  __shared__ _Float16 sh1bn [16 * H1D];   // bn1(h1_t), f16 (LSTM2 input)
  __shared__ _Float16 sh2   [16 * H2D];   // h2_t (raw), f16
  __shared__ _Float16 sh2bn [16 * H2D];   // bn2(h2_t), f16 (dense input)
  __shared__ float    sredm[8 * 16];      // per-wave row max partials
  __shared__ float    sreds[8 * 16];      // per-wave row sum partials
  __shared__ float    srowm[16];          // final per-row max
  __shared__ float    srows[16];          // final per-row 1/sum

  const int tid  = threadIdx.x;
  const int wave = tid >> 5;
  const int lane = tid & 31;
  const int col  = lane & 15;             // N / row index within tile
  const int rhi  = (lane >> 4) << 3;      // +8 rows for high half-lanes
  const int b0   = blockIdx.x * 16;       // batch-row base of this workgroup

  // zero recurrent state
  for (int i = tid; i < 16 * H1D; i += 256) { sh1raw[i] = (_Float16)0; sh1bn[i] = (_Float16)0; }
  for (int i = tid; i < 16 * H2D; i += 256) { sh2[i]    = (_Float16)0; sh2bn[i] = (_Float16)0; }
  __syncthreads();

  // per-lane constants, LSTM1 (wave w owns hidden1 units 16w..16w+15)
  const int hid1 = wave * 16 + col;
  float bia1[4];
#pragma unroll
  for (int g = 0; g < 4; ++g) bia1[g] = b1[g * H1D + hid1];
  const float sc1 = g1[hid1] * rsqrtf(v1[hid1] + EPSB);
  const float sf1 = beta1[hid1] - m1[hid1] * sc1;

  // per-lane constants, LSTM2 (waves 0..3 own hidden2 units 16w..16w+15)
  float bia2[4] = {0.f, 0.f, 0.f, 0.f};
  float sc2 = 0.f, sf2 = 0.f;
  int hid2 = 0;
  if (wave < 4) {
    hid2 = wave * 16 + col;
#pragma unroll
    for (int g = 0; g < 4; ++g) bia2[g] = b2[g * H2D + hid2];
    sc2 = g2[hid2] * rsqrtf(v2[hid2] + EPSB);
    sf2 = beta2[hid2] - m2[hid2] * sc2;
  }

  v8f c1 = splat8(0.f);                   // LSTM1 cell state (16x16 slice)
  v8f c2 = splat8(0.f);                   // LSTM2 cell state

  const int t1[4] = {wave, wave + 8, wave + 16, wave + 24};  // LSTM1 i/f/g/o tiles
  const int t2[4] = {wave, wave + 4, wave + 8, wave + 12};   // LSTM2 i/f/g/o tiles

  for (int t = 0; t < TLEN; ++t) {
    __syncthreads();  // b0: prior-step LDS writes visible

    // ---- phase A: load A-fragments of t-1 state + embeddings for step t ----
    const int id = ids[(b0 + col) * TLEN + t];
    const _Float16* ep = emb16 + (size_t)id * EDIM + ((lane < 16) ? 0 : 8);
    v8h xlo = *(const v8h*)ep;
    v8h xhi = *(const v8h*)(ep + 16);
    const v16h ax = cat8(xlo, xhi);

    v16h ah1[4];
#pragma unroll
    for (int c = 0; c < 4; ++c) ah1[c] = load_a(sh1raw, H1D, col, c * 32, lane);

    v16h ah2[2];
    if (wave < 4) {
#pragma unroll
      for (int c = 0; c < 2; ++c) ah2[c] = load_a(sh2, H2D, col, c * 32, lane);
    }
    __syncthreads();  // b1: state reads complete before overwrite

    // ---- phase B: LSTM1  z = x@W1 + h1@U1 + b1 -> gates -> c1,h1 ----
    v8f acc1[4];
#pragma unroll
    for (int g = 0; g < 4; ++g) {
      v8f a = splat8(bia1[g]);
      a = wmma_f16(ax, load_b(w1f, t1[g], 0, 1, lane), a);
#pragma unroll
      for (int c = 0; c < 4; ++c)
        a = wmma_f16(ah1[c], load_b(u1f, t1[g], c, 4, lane), a);
      acc1[g] = a;
    }
#pragma unroll
    for (int r = 0; r < 8; ++r) {
      float iv = sigmoidf(acc1[0][r]);
      float fv = sigmoidf(acc1[1][r]);
      float gv = fmaxf(acc1[2][r], 0.f);       // activation = relu
      float ov = sigmoidf(acc1[3][r]);
      float cc = fv * c1[r] + iv * gv;
      c1[r] = cc;
      float hv = ov * fmaxf(cc, 0.f);          // h = o * relu(c)
      int row = r + rhi;
      sh1raw[row * H1D + hid1] = (_Float16)hv;
      sh1bn [row * H1D + hid1] = (_Float16)(sc1 * hv + sf1);
    }
    __syncthreads();  // b2: bn1(h1_t) ready for LSTM2

    // ---- phase C: LSTM2 (waves 0..3; uniform branch -> execz skip) ----
    if (wave < 4) {
      v16h ahb[4];
#pragma unroll
      for (int c = 0; c < 4; ++c) ahb[c] = load_a(sh1bn, H1D, col, c * 32, lane);

      v8f acc2[4];
#pragma unroll
      for (int g = 0; g < 4; ++g) {
        v8f a = splat8(bia2[g]);
#pragma unroll
        for (int c = 0; c < 4; ++c)
          a = wmma_f16(ahb[c], load_b(w2f, t2[g], c, 4, lane), a);
#pragma unroll
        for (int c = 0; c < 2; ++c)
          a = wmma_f16(ah2[c], load_b(u2f, t2[g], c, 2, lane), a);
        acc2[g] = a;
      }
#pragma unroll
      for (int r = 0; r < 8; ++r) {
        float iv = sigmoidf(acc2[0][r]);
        float fv = sigmoidf(acc2[1][r]);
        float gv = fmaxf(acc2[2][r], 0.f);
        float ov = sigmoidf(acc2[3][r]);
        float cc = fv * c2[r] + iv * gv;
        c2[r] = cc;
        float hv = ov * fmaxf(cc, 0.f);
        int row = r + rhi;
        sh2  [row * H2D + hid2] = (_Float16)hv;
        sh2bn[row * H2D + hid2] = (_Float16)(sc2 * hv + sf2);
      }
    }
  }
  __syncthreads();  // final bn2(h2_T) ready

  // ---- Dense head:  logits[16,32000] = bn2(h2) @ Wd + bd ----
  // Pass 1: WMMA logits -> global, maintain ONLINE per-row (max, sum).
  v16h ad[2];
  ad[0] = load_a(sh2bn, H2D, col, 0, lane);
  ad[1] = load_a(sh2bn, H2D, col, 32, lane);

  const int nTiles = VDICT / 16;  // 2000
  float rm[8], rs[8];
#pragma unroll
  for (int r = 0; r < 8; ++r) { rm[r] = -3.4e38f; rs[r] = 0.f; }

  for (int n = wave; n < nTiles; n += 8) {
    // prefetch next tile's B-fragment records (both K-chunks) into caches
    __builtin_prefetch(wdf + (size_t)(((n + 8) * 2 + 0) * 32 + lane) * 16, 0, 3);
    __builtin_prefetch(wdf + (size_t)(((n + 8) * 2 + 1) * 32 + lane) * 16, 0, 3);

    v8f acc = splat8(bd[n * 16 + col]);
    acc = wmma_f16(ad[0], load_b(wdf, n, 0, 2, lane), acc);
    acc = wmma_f16(ad[1], load_b(wdf, n, 1, 2, lane), acc);
#pragma unroll
    for (int r = 0; r < 8; ++r) {
      int row = r + rhi;
      float v = acc[r];
      out[(size_t)(b0 + row) * VDICT + n * 16 + col] = v;
      float nm = fmaxf(rm[r], v);                       // online max/sum update
      rs[r] = rs[r] * __expf(rm[r] - nm) + __expf(v - nm);
      rm[r] = nm;
    }
  }
  // combine (max,sum) pairs across the 16 lanes of each half-wave
#pragma unroll
  for (int msk = 8; msk >= 1; msk >>= 1) {
#pragma unroll
    for (int r = 0; r < 8; ++r) {
      float om = __shfl_xor(rm[r], msk, 32);
      float os = __shfl_xor(rs[r], msk, 32);
      float nm = fmaxf(rm[r], om);
      rs[r] = rs[r] * __expf(rm[r] - nm) + os * __expf(om - nm);
      rm[r] = nm;
    }
  }
  if (lane == 0) {
#pragma unroll
    for (int r = 0; r < 8; ++r) { sredm[wave * 16 + r] = rm[r]; sreds[wave * 16 + r] = rs[r]; }
  }
  if (lane == 16) {
#pragma unroll
    for (int r = 0; r < 8; ++r) { sredm[wave * 16 + 8 + r] = rm[r]; sreds[wave * 16 + 8 + r] = rs[r]; }
  }
  __syncthreads();
  if (tid < 16) {   // combine the 8 wave partials per row
    float M = sredm[tid], S = sreds[tid];
    for (int w = 1; w < 8; ++w) {
      float om = sredm[w * 16 + tid], os = sreds[w * 16 + tid];
      float nm = fmaxf(M, om);
      S = S * __expf(M - nm) + os * __expf(om - nm);
      M = nm;
    }
    srowm[tid] = M;
    srows[tid] = 1.0f / S;
  }
  __syncthreads();
  float rowm[8], rinv[8];
#pragma unroll
  for (int r = 0; r < 8; ++r) { rowm[r] = srowm[r + rhi]; rinv[r] = srows[r + rhi]; }

  // Pass 2: single normalize pass  out = exp(x - max) / sum
  for (int n = wave; n < nTiles; n += 8) {
#pragma unroll
    for (int r = 0; r < 8; ++r) {
      int row = r + rhi;
      size_t idx = (size_t)(b0 + row) * VDICT + n * 16 + col;
      out[idx] = __expf(out[idx] - rowm[r]) * rinv[r];
    }
  }
}

// ---------------------------------------------------------------------------
extern "C" void kernel_launch(void* const* d_in, const int* in_sizes, int n_in,
                              void* d_out, int out_size, void* d_ws, size_t ws_size,
                              hipStream_t stream) {
  (void)in_sizes; (void)n_in; (void)out_size; (void)ws_size;
  const int*   ids   = (const int*)  d_in[0];
  const float* emb   = (const float*)d_in[1];
  const float* w1    = (const float*)d_in[2];
  const float* u1    = (const float*)d_in[3];
  const float* b1    = (const float*)d_in[4];
  const float* g1    = (const float*)d_in[5];
  const float* beta1 = (const float*)d_in[6];
  const float* m1    = (const float*)d_in[7];
  const float* v1    = (const float*)d_in[8];
  const float* w2    = (const float*)d_in[9];
  const float* u2    = (const float*)d_in[10];
  const float* b2    = (const float*)d_in[11];
  const float* g2    = (const float*)d_in[12];
  const float* beta2 = (const float*)d_in[13];
  const float* m2    = (const float*)d_in[14];
  const float* v2    = (const float*)d_in[15];
  const float* wd    = (const float*)d_in[16];
  const float* bd    = (const float*)d_in[17];
  float* out = (float*)d_out;

  // workspace layout (f16 elements), all 32B-aligned sections
  _Float16* ws    = (_Float16*)d_ws;
  _Float16* emb16 = ws;                               // 32001*32       = 1,024,032
  _Float16* w1f   = emb16 + 1024032;                  // 32t*1c*512     =    16,384
  _Float16* u1f   = w1f   + 32 * 1 * 512;             // 32t*4c*512     =    65,536
  _Float16* w2f   = u1f   + 32 * 4 * 512;             // 16t*4c*512     =    32,768
  _Float16* u2f   = w2f   + 16 * 4 * 512;             // 16t*2c*512     =    16,384
  _Float16* wdf   = u2f   + 16 * 2 * 512;             // 2000t*2c*512   = 2,048,000
  // total ~6.2 MB of d_ws

  const int nEmb = 32001 * 32;
  cvt_half_kernel<<<(nEmb + 255) / 256, 256, 0, stream>>>(emb, emb16, nEmb);
  pack_b_kernel<<<(32 * 1 * 32 + 255) / 256, 256, 0, stream>>>(w1, w1f, 32, 1, 4 * H1D);
  pack_b_kernel<<<(32 * 4 * 32 + 255) / 256, 256, 0, stream>>>(u1, u1f, 32, 4, 4 * H1D);
  pack_b_kernel<<<(16 * 4 * 32 + 255) / 256, 256, 0, stream>>>(w2, w2f, 16, 4, 4 * H2D);
  pack_b_kernel<<<(16 * 2 * 32 + 255) / 256, 256, 0, stream>>>(u2, u2f, 16, 2, 4 * H2D);
  pack_b_kernel<<<(2000 * 2 * 32 + 255) / 256, 256, 0, stream>>>(wd, wdf, 2000, 2, VDICT);

  lstm_fused_kernel<<<512 / 16, 256, 0, stream>>>(
      ids, emb16, w1f, u1f, w2f, u2f, wdf,
      b1, g1, beta1, m1, v1,
      b2, g2, beta2, m2, v2,
      bd, out);
}
